// PNANetNS_83133386981990
// MI455X (gfx1250) — matrix operand order repaired and checked by reading, not compiled, not run on hardware
//
#include <hip/hip_runtime.h>
#include <hip/hip_bf16.h>
#include <math.h>

#define EPS_MSG 1e-7f
#define EPS_SM  1e-16f

typedef _Float16 h8_t  __attribute__((ext_vector_type(8)));
typedef _Float16 h16_t __attribute__((ext_vector_type(16)));
typedef float    f8_t  __attribute__((ext_vector_type(8)));

// ---------------- problem dims (fixed by setup_inputs) ----------------
#define N0   200000
#define N1   50000
#define N2   5000
#define N2P  5008      // 5000 padded to multiple of 16
#define E0   500000
#define E1   100000
#define FIN  100
#define FINP 128       // K-pad of 100
#define HH   128
#define H2   256
#define C1   47
#define C1NP 48        // N-pad of 47
#define C1KP 64        // K-pad of 47
#define C2   94
#define C2P  96        // pad of 94

// ---------------- elementwise helpers ----------------
__global__ __launch_bounds__(256) void fill_f32(float* p, float v, long long n) {
    long long i = (long long)blockIdx.x * blockDim.x + threadIdx.x;
    if (i < n) p[i] = v;
}

// f32 [Mreal x Kreal (ld=ldsrc)] -> f16 [Mpad x Kpad], zero padded
__global__ __launch_bounds__(256) void cast_pad_f16(const float* __restrict__ src,
                                                    _Float16* __restrict__ dst,
                                                    int Mreal, int ldsrc, int Kreal,
                                                    long long Mpad, int Kpad) {
    long long i = (long long)blockIdx.x * blockDim.x + threadIdx.x;
    long long n = Mpad * (long long)Kpad;
    if (i >= n) return;
    long long r = i / Kpad;
    int k = (int)(i % Kpad);
    float v = (r < (long long)Mreal && k < Kreal) ? src[r * (long long)ldsrc + k] : 0.0f;
    dst[i] = (_Float16)v;
}

// W [K x N] f32 row-major -> Bt [Npad x Kpad] f16 (B operand layout = N-major)
__global__ __launch_bounds__(256) void transpose_pad_f16(const float* __restrict__ W,
                                                         _Float16* __restrict__ Bt,
                                                         int K, int N, int Kpad, int Npad) {
    long long i = (long long)blockIdx.x * blockDim.x + threadIdx.x;
    long long n = (long long)Npad * Kpad;
    if (i >= n) return;
    int nn = (int)(i / Kpad);
    int k  = (int)(i % Kpad);
    float v = (nn < N && k < K) ? W[(long long)k * N + nn] : 0.0f;
    Bt[i] = (_Float16)v;
}

// ---------------- WMMA GEMM: C[M,N] = act(A[M,K] @ B[K,N] + bias) ----------------
// A: f16 [Mtiles*16 x Kpad] row-major.  Bt: f16 [Ntiles*16 x Kpad] (pre-transposed B).
// One 16x(16*NB) output strip per wave: A loaded once per k-step, reused for NB WMMAs.
template <int NB>
__global__ __launch_bounds__(256) void wmma_gemm_nb(
    const _Float16* __restrict__ A, const _Float16* __restrict__ Bt,
    const float* __restrict__ bias, float* __restrict__ C,
    int Mtiles, int NtGroups, int Kpad, int Nreal, int ldc, int act) {
    int wid = (int)((blockIdx.x * blockDim.x + threadIdx.x) >> 5);
    if (wid >= Mtiles * NtGroups) return;         // uniform per wave -> EXEC all-ones
    int mt = wid / NtGroups, ng = wid % NtGroups;
    int lane = threadIdx.x & 31;
    int lr = lane & 15;
    int hi = (lane >> 4) & 1;

    // A operand: halves 0-15 -> K = hi*16 .. hi*16+15, row = lr
    const _Float16* arow = A + (size_t)(mt * 16 + lr) * Kpad + hi * 16;
    // B operand (N x K): halves 0-15 -> K = hi*16 .. hi*16+15, column = lr
    const _Float16* brow[NB];
#pragma unroll
    for (int j = 0; j < NB; ++j)
        brow[j] = Bt + (size_t)((ng * NB + j) * 16 + lr) * Kpad + hi * 16;

    f8_t acc[NB];
#pragma unroll
    for (int j = 0; j < NB; ++j) acc[j] = (f8_t){0.f, 0.f, 0.f, 0.f, 0.f, 0.f, 0.f, 0.f};

    for (int k = 0; k < Kpad; k += 32) {
        h8_t a0 = *(const h8_t*)(arow + k);
        h8_t a1 = *(const h8_t*)(arow + k + 8);
        h16_t av;
#pragma unroll
        for (int q = 0; q < 8; ++q) { av[q] = a0[q]; av[q + 8] = a1[q]; }
#pragma unroll
        for (int j = 0; j < NB; ++j) {
            h16_t bv = *(const h16_t*)(brow[j] + k);
            acc[j] = __builtin_amdgcn_wmma_f32_16x16x32_f16(
                /*neg_a=*/false, av, /*neg_b=*/false, bv,
                /*c_mod=*/(short)0, acc[j], /*reuse_a=*/false, /*reuse_b=*/false);
        }
    }

    int r0 = mt * 16 + hi * 8;
#pragma unroll
    for (int j = 0; j < NB; ++j) {
        int col = (ng * NB + j) * 16 + lr;
        if (col < Nreal) {
            float bvl = bias ? bias[col] : 0.0f;
#pragma unroll
            for (int r = 0; r < 8; ++r) {
                float v = acc[j][r] + bvl;
                if (act == 1) v = fmaxf(v, 0.0f) + EPS_MSG;   // GENConv message relu + eps
                C[(size_t)(r0 + r) * ldc + col] = v;
            }
        }
    }
}

// ---------------- segment softmax (atomic 3-pass) ----------------
__device__ inline void atomicMaxF32(float* addr, float val) {
    unsigned int* ua = (unsigned int*)addr;
    unsigned int old = *ua;
    while (__int_as_float((int)old) < val) {
        unsigned int assumed = old;
        old = atomicCAS(ua, assumed, (unsigned int)__float_as_int(val));
        if (old == assumed) break;
    }
}

// vectorized (4 channels / thread) — layer 0, C multiple of 4
__global__ __launch_bounds__(256) void edge_seg_max4(const int* __restrict__ src,
                                                     const int* __restrict__ dst,
                                                     const float* __restrict__ m,
                                                     const float* __restrict__ tptr,
                                                     float* __restrict__ mx,
                                                     long long E, int Cv, int ld) {
    long long i = (long long)blockIdx.x * blockDim.x + threadIdx.x;
    if (i >= E * Cv) return;
    int e = (int)(i / Cv), c = (int)(i % Cv) * 4;
    float t = *tptr;
    float4 v = *(const float4*)(m + (size_t)src[e] * ld + c);
    float* p = mx + (size_t)dst[e] * ld + c;
    atomicMaxF32(p + 0, v.x * t);
    atomicMaxF32(p + 1, v.y * t);
    atomicMaxF32(p + 2, v.z * t);
    atomicMaxF32(p + 3, v.w * t);
}

__global__ __launch_bounds__(256) void edge_seg_acc4(const int* __restrict__ src,
                                                     const int* __restrict__ dst,
                                                     const float* __restrict__ m,
                                                     const float* __restrict__ tptr,
                                                     const float* __restrict__ mx,
                                                     float* __restrict__ den,
                                                     float* __restrict__ num,
                                                     long long E, int Cv, int ld) {
    long long i = (long long)blockIdx.x * blockDim.x + threadIdx.x;
    if (i >= E * Cv) return;
    int e = (int)(i / Cv), c = (int)(i % Cv) * 4;
    float t = *tptr;
    size_t si = (size_t)src[e] * ld + c;
    size_t di = (size_t)dst[e] * ld + c;
    float4 mv = *(const float4*)(m + si);
    float4 mxv = *(const float4*)(mx + di);
    float ex;
    ex = expf(mv.x * t - mxv.x); atomicAdd(&den[di + 0], ex); atomicAdd(&num[di + 0], mv.x * ex);
    ex = expf(mv.y * t - mxv.y); atomicAdd(&den[di + 1], ex); atomicAdd(&num[di + 1], mv.y * ex);
    ex = expf(mv.z * t - mxv.z); atomicAdd(&den[di + 2], ex); atomicAdd(&num[di + 2], mv.z * ex);
    ex = expf(mv.w * t - mxv.w); atomicAdd(&den[di + 3], ex); atomicAdd(&num[di + 3], mv.w * ex);
}

// scalar versions — layer 1 (C = 47)
__global__ __launch_bounds__(256) void edge_seg_max(const int* __restrict__ src,
                                                    const int* __restrict__ dst,
                                                    const float* __restrict__ m,
                                                    const float* __restrict__ tptr,
                                                    float* __restrict__ mx,
                                                    long long E, int C, int ld) {
    long long i = (long long)blockIdx.x * blockDim.x + threadIdx.x;
    if (i >= E * C) return;
    int e = (int)(i / C), c = (int)(i % C);
    float t = *tptr;
    float v = m[(size_t)src[e] * ld + c] * t;
    atomicMaxF32(&mx[(size_t)dst[e] * ld + c], v);
}

__global__ __launch_bounds__(256) void edge_seg_acc(const int* __restrict__ src,
                                                    const int* __restrict__ dst,
                                                    const float* __restrict__ m,
                                                    const float* __restrict__ tptr,
                                                    const float* __restrict__ mx,
                                                    float* __restrict__ den,
                                                    float* __restrict__ num,
                                                    long long E, int C, int ld) {
    long long i = (long long)blockIdx.x * blockDim.x + threadIdx.x;
    if (i >= E * C) return;
    int e = (int)(i / C), c = (int)(i % C);
    float t = *tptr;
    size_t si = (size_t)src[e] * ld + c;
    size_t di = (size_t)dst[e] * ld + c;
    float mv = m[si];
    float ex = expf(mv * t - mx[di]);
    atomicAdd(&den[di], ex);
    atomicAdd(&num[di], mv * ex);
}

__global__ __launch_bounds__(256) void fix_mx(float* mx, long long n) {
    long long i = (long long)blockIdx.x * blockDim.x + threadIdx.x;
    if (i >= n) return;
    float v = mx[i];
    if (!(v >= -3.0e38f)) mx[i] = 0.0f;    // -inf (empty segment) or NaN -> 0
}

// out[i] = num/den + out[i]   (den pre-initialized to EPS_SM so den = sum + eps)
__global__ __launch_bounds__(256) void seg_combine(const float* __restrict__ num,
                                                   const float* __restrict__ den,
                                                   float* __restrict__ out,
                                                   long long M, int Nreal, int ld) {
    long long i = (long long)blockIdx.x * blockDim.x + threadIdx.x;
    if (i >= M * Nreal) return;
    long long r = i / Nreal;
    int c = (int)(i % Nreal);
    size_t idx = (size_t)r * ld + c;
    out[idx] = num[idx] / den[idx] + out[idx];
}

// ---------------- LayerNorm (+ReLU / +exact GELU), one wave per row ----------------
__global__ __launch_bounds__(256) void ln_act_kernel(float* __restrict__ x,
                                                     const float* __restrict__ g,
                                                     const float* __restrict__ b,
                                                     int M, int W, int ld, int act) {
    int row = blockIdx.x * 8 + (threadIdx.x >> 5);
    int lane = threadIdx.x & 31;
    if (row >= M) return;                          // uniform per wave
    float* p = x + (size_t)row * ld;
    float s = 0.f, ss = 0.f;
    for (int c = lane; c < W; c += 32) { float v = p[c]; s += v; ss += v * v; }
    for (int m = 16; m; m >>= 1) { s += __shfl_xor(s, m, 32); ss += __shfl_xor(ss, m, 32); }
    float inv = 1.0f / (float)W;
    float mu = s * inv;
    float var = ss * inv - mu * mu;
    float rstd = rsqrtf(var + 1e-5f);
    for (int c = lane; c < W; c += 32) {
        float v = (p[c] - mu) * rstd * g[c] + b[c];
        if (act == 1) v = fmaxf(v, 0.0f);
        else          v = 0.5f * v * (1.0f + erff(v * 0.70710678118654752f));  // exact GELU
        p[c] = v;
    }
}

// ---------------- log_softmax over last dim, one wave per row ----------------
__global__ __launch_bounds__(256) void log_softmax_kernel(const float* __restrict__ x,
                                                          float* __restrict__ out,
                                                          int M, int W, int ld) {
    int row = blockIdx.x * 8 + (threadIdx.x >> 5);
    int lane = threadIdx.x & 31;
    if (row >= M) return;
    const float* p = x + (size_t)row * ld;
    float mx = -3.4e38f;
    for (int c = lane; c < W; c += 32) mx = fmaxf(mx, p[c]);
    for (int m = 16; m; m >>= 1) mx = fmaxf(mx, __shfl_xor(mx, m, 32));
    float s = 0.f;
    for (int c = lane; c < W; c += 32) s += expf(p[c] - mx);
    for (int m = 16; m; m >>= 1) s += __shfl_xor(s, m, 32);
    float lse = mx + logf(s);
    for (int c = lane; c < W; c += 32) out[(size_t)row * W + c] = p[c] - lse;
}

// ---------------- launcher ----------------
extern "C" void kernel_launch(void* const* d_in, const int* in_sizes, int n_in,
                              void* d_out, int out_size, void* d_ws, size_t ws_size,
                              hipStream_t stream) {
    const float* x    = (const float*)d_in[0];
    const int*   ei0  = (const int*)d_in[1];   // [2, E0]: src then dst
    const int*   ei1  = (const int*)d_in[2];   // [2, E1]
    const float* t0   = (const float*)d_in[5];
    const float* Ws0  = (const float*)d_in[6];  const float* bs0 = (const float*)d_in[7];
    const float* Wd0  = (const float*)d_in[8];  const float* bd0 = (const float*)d_in[9];
    const float* W10  = (const float*)d_in[10]; const float* b10 = (const float*)d_in[11];
    const float* g10  = (const float*)d_in[12]; const float* be10= (const float*)d_in[13];
    const float* W20  = (const float*)d_in[14]; const float* b20 = (const float*)d_in[15];
    const float* ng0  = (const float*)d_in[16]; const float* nb0 = (const float*)d_in[17];
    const float* t1   = (const float*)d_in[18];
    const float* Ws1  = (const float*)d_in[19]; const float* bs1 = (const float*)d_in[20];
    const float* Wd1  = (const float*)d_in[21]; const float* bd1 = (const float*)d_in[22];
    const float* W11  = (const float*)d_in[23]; const float* b11 = (const float*)d_in[24];
    const float* g11  = (const float*)d_in[25]; const float* be11= (const float*)d_in[26];
    const float* W21  = (const float*)d_in[27]; const float* b21 = (const float*)d_in[28];
    float* out = (float*)d_out;

    // ---- workspace bump allocator (256B aligned), with phase-safe aliasing ----
    char* ws = (char*)d_ws;
    size_t off = 0;
    auto alloc = [&](size_t bytes) -> void* {
        off = (off + 255) & ~(size_t)255;
        void* p = ws + off;
        off += bytes;
        return p;
    };
    _Float16* Xh   = (_Float16*)alloc((size_t)N0 * FINP * 2);   // 51.2MB, reused as h2a
    float*    h2a  = (float*)Xh;                                // 50000*256*4 == same bytes
    _Float16* WS0t = (_Float16*)alloc((size_t)HH * FINP * 2);
    _Float16* WD0t = (_Float16*)alloc((size_t)HH * FINP * 2);
    _Float16* W10t = (_Float16*)alloc((size_t)H2 * HH * 2);
    _Float16* W20t = (_Float16*)alloc((size_t)HH * H2 * 2);
    _Float16* WS1t = (_Float16*)alloc((size_t)C1NP * HH * 2);
    _Float16* WD1t = (_Float16*)alloc((size_t)C1NP * HH * 2);
    _Float16* W11t = (_Float16*)alloc((size_t)C2P * C1KP * 2);
    _Float16* W21t = (_Float16*)alloc((size_t)C1NP * C2P * 2);
    float* m0   = (float*)alloc((size_t)N0 * HH * 4);           // 102.4MB node messages
    float* D0   = (float*)alloc((size_t)N1 * HH * 4);           // lin_dst out, becomes out0
    float* mx0  = (float*)alloc((size_t)N1 * HH * 4);           // reused as x1
    float* x1   = mx0;
    float* den0 = (float*)alloc((size_t)N1 * HH * 4);           // reused as h2h (f16, half bytes)
    _Float16* h2h = (_Float16*)den0;
    float* num0 = (float*)alloc((size_t)N1 * HH * 4);           // reused as x1h + m1
    _Float16* x1h = (_Float16*)num0;
    float* m1   = (float*)((char*)num0 + (size_t)13 * 1024 * 1024);
    _Float16* out0h = (_Float16*)alloc((size_t)N1 * HH * 2);
    float* D1   = (float*)alloc((size_t)N2P * C1NP * 4);
    float* mx1  = (float*)alloc((size_t)N2P * C1NP * 4);
    float* den1 = (float*)alloc((size_t)N2P * C1NP * 4);
    float* num1 = (float*)alloc((size_t)N2P * C1NP * 4);
    _Float16* out1h = (_Float16*)alloc((size_t)N2P * C1KP * 2);
    float* h2b  = (float*)alloc((size_t)N2P * C2P * 4);
    _Float16* h2bh = (_Float16*)alloc((size_t)N2P * C2P * 2);
    float* x2   = (float*)alloc((size_t)N2P * C1NP * 4);

    auto blocks = [](long long n) { return (unsigned)((n + 255) / 256); };
    // pick NB=4 when Ntiles%4==0 else NB=3 (all shapes here are 8,16,3,6)
    auto gemm = [&](const _Float16* A, const _Float16* Bt, const float* bias, float* C,
                    int Mt, int Ntiles, int Kpad, int Nreal, int ldc, int act) {
        if ((Ntiles & 3) == 0) {
            int ng = Ntiles / 4, groups = Mt * ng;
            wmma_gemm_nb<4><<<(groups + 7) / 8, 256, 0, stream>>>(A, Bt, bias, C, Mt, ng,
                                                                  Kpad, Nreal, ldc, act);
        } else {
            int ng = Ntiles / 3, groups = Mt * ng;
            wmma_gemm_nb<3><<<(groups + 7) / 8, 256, 0, stream>>>(A, Bt, bias, C, Mt, ng,
                                                                  Kpad, Nreal, ldc, act);
        }
    };

    // ---- phase 0: conversions ----
    cast_pad_f16<<<blocks((long long)N0 * FINP), 256, 0, stream>>>(x, Xh, N0, FIN, FIN, N0, FINP);
    transpose_pad_f16<<<blocks(HH * FINP), 256, 0, stream>>>(Ws0, WS0t, FIN, HH, FINP, HH);
    transpose_pad_f16<<<blocks(HH * FINP), 256, 0, stream>>>(Wd0, WD0t, FIN, HH, FINP, HH);
    transpose_pad_f16<<<blocks(H2 * HH), 256, 0, stream>>>(W10, W10t, HH, H2, HH, H2);
    transpose_pad_f16<<<blocks(HH * H2), 256, 0, stream>>>(W20, W20t, H2, HH, H2, HH);
    transpose_pad_f16<<<blocks(C1NP * HH), 256, 0, stream>>>(Ws1, WS1t, HH, C1, HH, C1NP);
    transpose_pad_f16<<<blocks(C1NP * HH), 256, 0, stream>>>(Wd1, WD1t, HH, C1, HH, C1NP);
    transpose_pad_f16<<<blocks(C2P * C1KP), 256, 0, stream>>>(W11, W11t, C1, C2, C1KP, C2P);
    transpose_pad_f16<<<blocks(C1NP * C2P), 256, 0, stream>>>(W21, W21t, C2, C1, C2P, C1NP);

    // ---- layer 0 ----
    gemm(Xh, WS0t, bs0, m0, N0 / 16, HH / 16, FINP, HH, HH, 1);   // m = relu(x@Ws+bs)+eps
    gemm(Xh, WD0t, bd0, D0, N1 / 16, HH / 16, FINP, HH, HH, 0);   // lin_dst

    long long n0e = (long long)N1 * HH;
    fill_f32<<<blocks(n0e), 256, 0, stream>>>(mx0, -__builtin_inff(), n0e);
    edge_seg_max4<<<blocks((long long)E0 * (HH / 4)), 256, 0, stream>>>(ei0, ei0 + E0, m0, t0, mx0, E0, HH / 4, HH);
    fix_mx<<<blocks(n0e), 256, 0, stream>>>(mx0, n0e);
    fill_f32<<<blocks(n0e), 256, 0, stream>>>(den0, EPS_SM, n0e);
    fill_f32<<<blocks(n0e), 256, 0, stream>>>(num0, 0.0f, n0e);
    edge_seg_acc4<<<blocks((long long)E0 * (HH / 4)), 256, 0, stream>>>(ei0, ei0 + E0, m0, t0, mx0, den0, num0, E0, HH / 4, HH);
    seg_combine<<<blocks(n0e), 256, 0, stream>>>(num0, den0, D0, N1, HH, HH);  // D0 = out

    cast_pad_f16<<<blocks((long long)N1 * HH), 256, 0, stream>>>(D0, out0h, N1, HH, HH, N1, HH);
    gemm(out0h, W10t, b10, h2a, N1 / 16, H2 / 16, HH, H2, H2, 0);              // MLP fc1
    ln_act_kernel<<<(N1 + 7) / 8, 256, 0, stream>>>(h2a, g10, be10, N1, H2, H2, 1);  // LN+ReLU
    cast_pad_f16<<<blocks((long long)N1 * H2), 256, 0, stream>>>(h2a, h2h, N1, H2, H2, N1, H2);
    gemm(h2h, W20t, b20, x1, N1 / 16, HH / 16, H2, HH, HH, 0);                 // MLP fc2
    ln_act_kernel<<<(N1 + 7) / 8, 256, 0, stream>>>(x1, ng0, nb0, N1, HH, HH, 2);   // LN+GELU
    cast_pad_f16<<<blocks((long long)N1 * HH), 256, 0, stream>>>(x1, x1h, N1, HH, HH, N1, HH);

    // ---- layer 1 ----
    gemm(x1h, WS1t, bs1, m1, N1 / 16, C1NP / 16, HH, C1, C1NP, 1);
    gemm(x1h, WD1t, bd1, D1, N2P / 16, C1NP / 16, HH, C1, C1NP, 0);

    long long n1e = (long long)N2P * C1NP;
    fill_f32<<<blocks(n1e), 256, 0, stream>>>(mx1, -__builtin_inff(), n1e);
    edge_seg_max<<<blocks((long long)E1 * C1), 256, 0, stream>>>(ei1, ei1 + E1, m1, t1, mx1, E1, C1, C1NP);
    fix_mx<<<blocks(n1e), 256, 0, stream>>>(mx1, n1e);
    fill_f32<<<blocks(n1e), 256, 0, stream>>>(den1, EPS_SM, n1e);
    fill_f32<<<blocks(n1e), 256, 0, stream>>>(num1, 0.0f, n1e);
    edge_seg_acc<<<blocks((long long)E1 * C1), 256, 0, stream>>>(ei1, ei1 + E1, m1, t1, mx1, den1, num1, E1, C1, C1NP);
    seg_combine<<<blocks((long long)N2 * C1), 256, 0, stream>>>(num1, den1, D1, N2, C1, C1NP);

    cast_pad_f16<<<blocks((long long)N2P * C1KP), 256, 0, stream>>>(D1, out1h, N2, C1NP, C1, N2P, C1KP);
    gemm(out1h, W11t, b11, h2b, N2P / 16, C2P / 16, C1KP, C2, C2P, 0);
    ln_act_kernel<<<(N2 + 7) / 8, 256, 0, stream>>>(h2b, g11, be11, N2, C2, C2P, 1);
    cast_pad_f16<<<blocks((long long)N2P * C2P), 256, 0, stream>>>(h2b, h2bh, N2, C2P, C2, N2P, C2P);
    gemm(h2bh, W21t, b21, x2, N2P / 16, C1NP / 16, C2P, C1, C1NP, 0);

    log_softmax_kernel<<<(N2 + 7) / 8, 256, 0, stream>>>(x2, out, N2, C1, C1NP);
}
